// HGNN_59742995088039
// MI455X (gfx1250) — matrix operand order repaired and checked by reading, not compiled, not run on hardware
//
#include <hip/hip_runtime.h>

#define N_NODES 1024
#define N_EDGES 524288
#define IN_C   64
#define HID_C  128
#define OUT_C  512
#define MLP_H  64

typedef __attribute__((ext_vector_type(16))) __bf16 v16bf;
typedef __attribute__((ext_vector_type(8)))  float  v8f;

union BF16x16 { unsigned short u[16]; v16bf v; };
union F32x8   { float f[8];           v8f   v; };

__device__ __forceinline__ unsigned short f32_to_bf16_bits(float f) {
    union { float f; unsigned int u; } x; x.f = f;
    unsigned int lsb = (x.u >> 16) & 1u;
    return (unsigned short)((x.u + 0x7FFFu + lsb) >> 16);
}

// ---------------------------------------------------------------- CSR build
__global__ void k_zero_i32(int* p, int n) {
    int i = blockIdx.x * blockDim.x + threadIdx.x;
    if (i < n) p[i] = 0;
}

__global__ void k_hist(const int* __restrict__ ei, int* __restrict__ deg) {
    int e = blockIdx.x * blockDim.x + threadIdx.x;
    if (e < N_EDGES) atomicAdd(&deg[ei[N_EDGES + e]], 1);
}

__global__ __launch_bounds__(1024) void k_scan(const int* __restrict__ deg,
                                               int* __restrict__ rowptr,
                                               int* __restrict__ cursor) {
    __shared__ int s[N_NODES];
    int t = threadIdx.x;
    int d = deg[t];
    s[t] = d;
    __syncthreads();
    #pragma unroll
    for (int off = 1; off < N_NODES; off <<= 1) {
        int v = (t >= off) ? s[t - off] : 0;
        __syncthreads();
        s[t] += v;
        __syncthreads();
    }
    if (t == 0) rowptr[0] = 0;
    rowptr[t + 1] = s[t];
    cursor[t]     = s[t] - d;   // exclusive
}

__global__ void k_scatter(const int* __restrict__ ei, int* __restrict__ cursor,
                          int* __restrict__ esrc) {
    int e = blockIdx.x * blockDim.x + threadIdx.x;
    if (e < N_EDGES) {
        int dst = ei[N_EDGES + e];
        int pos = atomicAdd(&cursor[dst], 1);
        esrc[pos] = ei[e];
    }
}

// ------------------------------------------------------------- converts
__global__ void k_f32_to_bf16(const float* __restrict__ in,
                              unsigned short* __restrict__ out, int n) {
    int i = blockIdx.x * blockDim.x + threadIdx.x;
    if (i < n) out[i] = f32_to_bf16_bits(in[i]);
}

// ----------------------------------------------- per-node partials A,B (bias in A)
// A[n][k] = b[k] + sum_c X[n][c] * W[c][k]        (dst half of W)
// B[n][k] =        sum_c X[n][c] * W[Cin+c][k]    (src half of W)
__global__ __launch_bounds__(64) void k_partial(const float* __restrict__ X, int Cin,
                                                const float* __restrict__ W,
                                                const float* __restrict__ b,
                                                float* __restrict__ A,
                                                float* __restrict__ B) {
    __shared__ float row[128];
    int n = blockIdx.x, k = threadIdx.x;
    for (int c = k; c < Cin; c += 64) row[c] = X[n * Cin + c];
    __syncthreads();
    float a = b[k], bb = 0.f;
    for (int c = 0; c < Cin; ++c) {
        float xc = row[c];
        a  += xc * W[c * MLP_H + k];
        bb += xc * W[(Cin + c) * MLP_H + k];
    }
    A[n * MLP_H + k] = a;
    B[n * MLP_H + k] = bb;
}

// ---------------------------------------------------------------- main WMMA kernel
// One block (8 wave32) per dst node. Edge tiles of 32 rows, double-buffered in LDS:
// build tile t+1 while WMMA-consuming tile t (one barrier per tile).
// [32x64]x[64xCOUT] via v_wmma_f32_16x16x32_bf16 (2 M sub-tiles x 2 chained K=32),
// elementwise running max over edge-row tiles held in registers.
#define TILE_M 32

__device__ __forceinline__ void build_tile(int t, int r0, int r1, int tid,
                                           const float* __restrict__ sA,
                                           const float* __restrict__ Bpart,
                                           const int* __restrict__ esrc,
                                           unsigned short* __restrict__ buf) {
    // 32 rows x 64 cols = 2048 elems / 256 threads = 8 per thread.
    // esrc reads are wave-uniform per 64-thread group -> broadcast from L0/L2.
    #pragma unroll
    for (int it = 0; it < 8; ++it) {
        int idx = tid + it * 256;
        int e = idx >> 6, k = idx & 63;
        int ei = r0 + t * TILE_M + e;
        int s = esrc[(ei < r1) ? ei : r0];   // pad with a real edge (max-idempotent)
        float v = sA[k] + Bpart[s * MLP_H + k];
        v = v > 0.f ? v : 0.f;
        buf[idx] = f32_to_bf16_bits(v);
    }
}

template <int COUT, int CT_PER_WAVE>
__global__ __launch_bounds__(256) void k_edge_wmma(
        const float* __restrict__ Apart,         // [N][64] (bias folded)
        const float* __restrict__ Bpart,         // [N][64]
        const unsigned short* __restrict__ Wbf,  // [64][COUT] bf16 bits
        const float* __restrict__ bias2,         // [COUT]
        const int* __restrict__ rowptr,
        const int* __restrict__ esrc,
        float* __restrict__ Hout) {              // [N][COUT]
    __shared__ float sA[MLP_H];
    __shared__ unsigned short shid[2][TILE_M * MLP_H];

    const int node = blockIdx.x;
    const int tid  = threadIdx.x;
    const int wave = tid >> 5;
    const int lane = tid & 31;
    const int half = lane >> 4;
    const int mrow = lane & 15;

    // resident B fragments for this wave's column tiles
    // B layout (16-bit, 32x16): lane N = lane%16; q<8: K = half*8 + q, q>=8: K = 16 + half*8 + (q-8)
    BF16x16 bfrag[CT_PER_WAVE][2];
    #pragma unroll
    for (int ct = 0; ct < CT_PER_WAVE; ++ct) {
        int ncol = (wave * CT_PER_WAVE + ct) * 16 + mrow;
        #pragma unroll
        for (int kh = 0; kh < 2; ++kh)
            #pragma unroll
            for (int q = 0; q < 16; ++q) {
                int K = kh * 32 + ((q < 8) ? (half * 8 + q) : (16 + half * 8 + (q - 8)));
                bfrag[ct][kh].u[q] = Wbf[K * COUT + ncol];
            }
    }

    const int r0 = rowptr[node], r1 = rowptr[node + 1];
    const int deg = r1 - r0;

    if (tid < MLP_H) sA[tid] = Apart[node * MLP_H + tid];
    __syncthreads();

    float acc[CT_PER_WAVE][8];
    #pragma unroll
    for (int ct = 0; ct < CT_PER_WAVE; ++ct)
        #pragma unroll
        for (int v = 0; v < 8; ++v) acc[ct][v] = -__builtin_inff();

    const int ntiles = (deg + TILE_M - 1) / TILE_M;
    if (ntiles > 0) build_tile(0, r0, r1, tid, sA, Bpart, esrc, shid[0]);

    for (int t = 0; t < ntiles; ++t) {
        __syncthreads();                     // shid[t&1] ready for everyone
        if (t + 1 < ntiles) {
            if (tid < TILE_M && (r0 + (t + 2) * TILE_M + tid) < r1)
                __builtin_prefetch(&esrc[r0 + (t + 2) * TILE_M + tid], 0, 1);
            build_tile(t + 1, r0, r1, tid, sA, Bpart, esrc, shid[(t + 1) & 1]);
        }

        const unsigned short* hb = shid[t & 1];
        // A fragments (16-bit 16x32): lane<16 row=mrow K 0-7,16-23; lane>=16 K 8-15,24-31
        BF16x16 af[2][2];                    // [M sub-tile][K half]
        #pragma unroll
        for (int sub = 0; sub < 2; ++sub)
            #pragma unroll
            for (int kh = 0; kh < 2; ++kh) {
                const unsigned short* hrow =
                    &hb[(sub * 16 + mrow) * MLP_H + kh * 32 + half * 8];
                #pragma unroll
                for (int q = 0; q < 8; ++q) af[sub][kh].u[q]     = hrow[q];
                #pragma unroll
                for (int q = 0; q < 8; ++q) af[sub][kh].u[8 + q] = hrow[16 + q];
            }

        #pragma unroll
        for (int ct = 0; ct < CT_PER_WAVE; ++ct)
            #pragma unroll
            for (int sub = 0; sub < 2; ++sub) {
                v8f d = {};
                d = __builtin_amdgcn_wmma_f32_16x16x32_bf16(
                        false, af[sub][0].v, false, bfrag[ct][0].v, (short)0, d, false, false);
                d = __builtin_amdgcn_wmma_f32_16x16x32_bf16(
                        false, af[sub][1].v, false, bfrag[ct][1].v, (short)0, d, false, false);
                F32x8 df; df.v = d;
                #pragma unroll
                for (int v = 0; v < 8; ++v) acc[ct][v] = fmaxf(acc[ct][v], df.f[v]);
            }
    }

    // reduce 16 edge-rows -> per-column max; D layout: VGPR v = M (v+8*half), N = lane%16
    #pragma unroll
    for (int ct = 0; ct < CT_PER_WAVE; ++ct) {
        float m = acc[ct][0];
        #pragma unroll
        for (int v = 1; v < 8; ++v) m = fmaxf(m, acc[ct][v]);
        m = fmaxf(m, __shfl_xor(m, 16, 32));
        if (half == 0) {
            int ncol = (wave * CT_PER_WAVE + ct) * 16 + mrow;
            Hout[node * COUT + ncol] = (deg == 0) ? 0.f : (m + bias2[ncol]);
        }
    }
}

// ---------------------------------------------------------------- readout GEMM
// out[m][n] = sum_k h2[k][m] * Wr[k][n] + br[n];  m,n in [0,512), k in [0,1024)
__global__ __launch_bounds__(256) void k_readout(const unsigned short* __restrict__ h2bf,
                                                 const unsigned short* __restrict__ wrbf,
                                                 const float* __restrict__ br,
                                                 float* __restrict__ out) {
    int wave = threadIdx.x >> 5, lane = threadIdx.x & 31;
    int half = lane >> 4, mrow = lane & 15;
    int tile = blockIdx.x * 8 + wave;       // 1024 tiles of 16x16
    int tm = tile >> 5, tn = tile & 31;

    v8f acc = {};
    for (int kt = 0; kt < 32; ++kt) {
        BF16x16 a, b;
        #pragma unroll
        for (int q = 0; q < 8; ++q) {
            int KaLo = kt * 32 + half * 8 + q;
            int KaHi = KaLo + 16;
            a.u[q]     = h2bf[KaLo * OUT_C + tm * 16 + mrow];
            a.u[8 + q] = h2bf[KaHi * OUT_C + tm * 16 + mrow];
        }
        #pragma unroll
        for (int q = 0; q < 16; ++q) {
            int Kb = kt * 32 + ((q < 8) ? (half * 8 + q) : (16 + half * 8 + (q - 8)));
            b.u[q] = wrbf[Kb * OUT_C + tn * 16 + mrow];
        }
        acc = __builtin_amdgcn_wmma_f32_16x16x32_bf16(false, a.v, false, b.v,
                                                      (short)0, acc, false, false);
    }
    F32x8 d; d.v = acc;
    float bv = br[tn * 16 + mrow];
    #pragma unroll
    for (int v = 0; v < 8; ++v) {
        int m = tm * 16 + half * 8 + v;
        out[m * OUT_C + tn * 16 + mrow] = d.f[v] + bv;
    }
}

// ---------------------------------------------------------------- launch
static inline size_t alignup(size_t x) { return (x + 255) & ~size_t(255); }

extern "C" void kernel_launch(void* const* d_in, const int* in_sizes, int n_in,
                              void* d_out, int out_size, void* d_ws, size_t ws_size,
                              hipStream_t stream) {
    const float* x  = (const float*)d_in[0];
    const int*   ei = (const int*)d_in[1];          // edge_index [2][E]
    const float* W1 = (const float*)d_in[2];
    const float* b1 = (const float*)d_in[3];
    const float* W2 = (const float*)d_in[4];
    const float* b2 = (const float*)d_in[5];
    const float* W3 = (const float*)d_in[6];
    const float* b3 = (const float*)d_in[7];
    const float* W4 = (const float*)d_in[8];
    const float* b4 = (const float*)d_in[9];
    const float* Wr = (const float*)d_in[10];
    const float* br = (const float*)d_in[11];
    float* out = (float*)d_out;
    (void)in_sizes; (void)n_in; (void)out_size; (void)ws_size;

    char* p = (char*)d_ws;
    size_t off = 0;
    auto take = [&](size_t bytes) { char* q = p + off; off += alignup(bytes); return q; };
    int* deg       = (int*)take(N_NODES * 4);
    int* rowptr    = (int*)take((N_NODES + 1) * 4);
    int* cursor    = (int*)take(N_NODES * 4);
    int* esrc      = (int*)take(N_EDGES * 4);
    float* A1      = (float*)take(N_NODES * MLP_H * 4);
    float* B1      = (float*)take(N_NODES * MLP_H * 4);
    float* h1      = (float*)take(N_NODES * HID_C * 4);
    float* A2      = (float*)take(N_NODES * MLP_H * 4);
    float* B2      = (float*)take(N_NODES * MLP_H * 4);
    float* h2      = (float*)take(N_NODES * OUT_C * 4);
    unsigned short* h2bf = (unsigned short*)take(N_NODES * OUT_C * 2);
    unsigned short* W2bf = (unsigned short*)take(MLP_H * HID_C * 2);
    unsigned short* W4bf = (unsigned short*)take(MLP_H * OUT_C * 2);
    unsigned short* Wrbf = (unsigned short*)take(N_NODES * OUT_C * 2);

    // CSR by dst
    k_zero_i32<<<4, 256, 0, stream>>>(deg, N_NODES);
    k_hist<<<N_EDGES / 256, 256, 0, stream>>>(ei, deg);
    k_scan<<<1, 1024, 0, stream>>>(deg, rowptr, cursor);
    k_scatter<<<N_EDGES / 256, 256, 0, stream>>>(ei, cursor, esrc);

    // weight conversion to bf16
    k_f32_to_bf16<<<(MLP_H * HID_C + 255) / 256, 256, 0, stream>>>(W2, W2bf, MLP_H * HID_C);
    k_f32_to_bf16<<<(MLP_H * OUT_C + 255) / 256, 256, 0, stream>>>(W4, W4bf, MLP_H * OUT_C);
    k_f32_to_bf16<<<(N_NODES * OUT_C + 255) / 256, 256, 0, stream>>>(Wr, Wrbf, N_NODES * OUT_C);

    // conv1
    k_partial<<<N_NODES, 64, 0, stream>>>(x, IN_C, W1, b1, A1, B1);
    k_edge_wmma<HID_C, 1><<<N_NODES, 256, 0, stream>>>(A1, B1, W2bf, b2, rowptr, esrc, h1);
    // conv2
    k_partial<<<N_NODES, 64, 0, stream>>>(h1, HID_C, W3, b3, A2, B2);
    k_edge_wmma<OUT_C, 4><<<N_NODES, 256, 0, stream>>>(A2, B2, W4bf, b4, rowptr, esrc, h2);
    // readout
    k_f32_to_bf16<<<(N_NODES * OUT_C + 255) / 256, 256, 0, stream>>>(h2, h2bf, N_NODES * OUT_C);
    k_readout<<<128, 256, 0, stream>>>(h2bf, Wrbf, br, out);
}
